// Inception_39685497815755
// MI455X (gfx1250) — 8-bit matrix operand order repaired and checked by reading, compiled, not run on hardware
//
#include <hip/hip_runtime.h>


typedef int v8i __attribute__((ext_vector_type(8)));
typedef signed char s8;

#define P_TOT 131072

__device__ __forceinline__ s8 q8(float x) {
    float r = rintf(x * 128.f);
    r = fminf(127.f, fmaxf(-127.f, r));
    return (s8)(int)r;
}

__global__ __launch_bounds__(256)
void quant_w_k(const float* __restrict__ w, s8* __restrict__ dst,
               int Co, int Ci, int KH, int Cpad, int total) {
    int i = blockIdx.x * 256 + threadIdx.x;
    if (i >= total) return;
    int c  = i % Cpad;
    int t2 = i / Cpad;
    int co = t2 % Co;
    int t  = t2 / Co;
    s8 v = 0;
    if (c < Ci) {
        int ky = t / KH, kx = t % KH;
        v = q8(w[(((size_t)co * Ci + c) * KH + ky) * KH + kx]);
    }
    dst[((size_t)t * Co + co) * Cpad + c] = v;
}

__global__ __launch_bounds__(256)
void quant_x_k(const float* __restrict__ x, s8* __restrict__ xq) {
    __shared__ __align__(16) s8 sm[32 * 192];
    const int p0  = blockIdx.x * 32;
    const int n   = p0 >> 10;
    const int hw0 = p0 & 1023;
    for (int e = threadIdx.x; e < 32 * 192; e += 256) {
        int c = e >> 5, i = e & 31;
        sm[i * 192 + c] = q8(x[((size_t)n * 192 + c) * 1024 + hw0 + i]);
    }
    __syncthreads();
    int* dst = (int*)(xq + (size_t)p0 * 192);
    for (int e = threadIdx.x; e < 1536; e += 256) dst[e] = ((const int*)sm)[e];
}

__global__ __launch_bounds__(256)
void pool_q_k(const float* __restrict__ x, s8* __restrict__ xpq) {
    __shared__ __align__(16) s8 sm[32 * 192];
    const int p0 = blockIdx.x * 32;
    const int n  = p0 >> 10;
    const int h  = (p0 >> 5) & 31;
    for (int e = threadIdx.x; e < 32 * 192; e += 256) {
        int c = e >> 5, w = e & 31;
        const float* bx = x + ((size_t)n * 192 + c) * 1024;
        float m = -3.4e38f;
        for (int dy = -1; dy <= 1; ++dy) {
            int hh = h + dy;
            if ((unsigned)hh < 32u)
                for (int dx = -1; dx <= 1; ++dx) {
                    int ww = w + dx;
                    if ((unsigned)ww < 32u) m = fmaxf(m, bx[hh * 32 + ww]);
                }
        }
        sm[w * 192 + c] = q8(m);
    }
    __syncthreads();
    int* dst = (int*)(xpq + (size_t)p0 * 192);
    for (int e = threadIdx.x; e < 1536; e += 256) dst[e] = ((const int*)sm)[e];
}

template <int TAPS, int CPAD>
__global__ __launch_bounds__(256, 2)
void conv_wmma_iu8(const s8* __restrict__ act, const s8* __restrict__ wq,
                   float* __restrict__ y, int Co) {
    const int lane = threadIdx.x & 31;
    const int wave = threadIdx.x >> 5;
    const int prow = blockIdx.x * 8 + wave;
    const int cot  = blockIdx.y;
    const int p0 = prow << 5;
    const int h  = (p0 >> 5) & 31;
    const int nn = lane & 15;
    const int hi = lane >> 4;
    const int kbA = hi << 4;
    const int kbB = hi << 4;

    v8i acc0 = {}, acc1 = {};
    const s8* wbase = wq + (size_t)(cot * 16 + nn) * CPAD;

#pragma unroll
    for (int t = 0; t < TAPS; ++t) {
        const int ky = (TAPS == 9) ? (t / 3 - 1) : 0;
        const int kx = (TAPS == 9) ? (t % 3 - 1) : 0;
        if ((unsigned)(h + ky) >= 32u) continue;
        const int msk0 = (kx < 0 && nn == 0)  ? 0 : -1;
        const int msk1 = (kx > 0 && nn == 15) ? 0 : -1;
        int pix0 = p0 + nn + ky * 32 + kx;
        int pix1 = pix0 + 16;
        pix0 = pix0 < 0 ? 0 : pix0;
        pix1 = pix1 >= P_TOT ? P_TOT - 1 : pix1;
        const s8* br0 = act + (size_t)pix0 * CPAD;
        const s8* br1 = act + (size_t)pix1 * CPAD;
        const s8* wt  = wbase + (size_t)t * Co * CPAD;

#pragma unroll
        for (int kc = 0; kc < CPAD; kc += 64) {
            int2 a0 = *(const int2*)(wt + kc + kbA);
            int2 a1 = *(const int2*)(wt + kc + kbA + 8);
            int2 a2 = *(const int2*)(wt + kc + kbA + 32);
            int2 a3 = *(const int2*)(wt + kc + kbA + 40);
            v8i Av;
            Av[0] = a0.x; Av[1] = a0.y; Av[2] = a1.x; Av[3] = a1.y;
            Av[4] = a2.x; Av[5] = a2.y; Av[6] = a3.x; Av[7] = a3.y;
            int4 b0 = *(const int4*)(br0 + kc + kbB);
            int4 b1 = *(const int4*)(br0 + kc + kbB + 32);
            int4 c0 = *(const int4*)(br1 + kc + kbB);
            int4 c1 = *(const int4*)(br1 + kc + kbB + 32);
            v8i Bv;
            Bv[0] = b0.x & msk0; Bv[1] = b0.y & msk0;
            Bv[2] = b0.z & msk0; Bv[3] = b0.w & msk0;
            Bv[4] = b1.x & msk0; Bv[5] = b1.y & msk0;
            Bv[6] = b1.z & msk0; Bv[7] = b1.w & msk0;
            acc0 = __builtin_amdgcn_wmma_i32_16x16x64_iu8(true, Av, true, Bv,
                                                          acc0, false, false);
            v8i Cv;
            Cv[0] = c0.x & msk1; Cv[1] = c0.y & msk1;
            Cv[2] = c0.z & msk1; Cv[3] = c0.w & msk1;
            Cv[4] = c1.x & msk1; Cv[5] = c1.y & msk1;
            Cv[6] = c1.z & msk1; Cv[7] = c1.w & msk1;
            acc1 = __builtin_amdgcn_wmma_i32_16x16x64_iu8(true, Av, true, Cv,
                                                          acc1, false, false);
        }
    }
    const float sc = 1.f / 16384.f;
    float* yo = y + (size_t)(cot * 16 + hi * 8) * P_TOT + p0 + nn;
#pragma unroll
    for (int r = 0; r < 8; ++r) {
        yo[(size_t)r * P_TOT]      = (float)acc0[r] * sc;
        yo[(size_t)r * P_TOT + 16] = (float)acc1[r] * sc;
    }
}

__global__ __launch_bounds__(256)
void bn_stats_k(const float* __restrict__ y, float* __restrict__ mean,
                float* __restrict__ rstd) {
    const int c = blockIdx.x;
    const float4* row = (const float4*)(y + (size_t)c * P_TOT);
    float s = 0.f, s2 = 0.f;
    for (int i = threadIdx.x; i < P_TOT / 4; i += 256) {
        float4 v = row[i];
        s  += v.x + v.y + v.z + v.w;
        s2 += v.x * v.x + v.y * v.y + v.z * v.z + v.w * v.w;
    }
    __shared__ float rs[256], rq[256];
    rs[threadIdx.x] = s; rq[threadIdx.x] = s2;
    __syncthreads();
    for (int o = 128; o > 0; o >>= 1) {
        if (threadIdx.x < o) {
            rs[threadIdx.x] += rs[threadIdx.x + o];
            rq[threadIdx.x] += rq[threadIdx.x + o];
        }
        __syncthreads();
    }
    if (threadIdx.x == 0) {
        float m = rs[0] * (1.f / P_TOT);
        float v = rq[0] * (1.f / P_TOT) - m * m;
        v = v > 0.f ? v : 0.f;
        mean[c] = m;
        rstd[c] = rsqrtf(v + 1e-5f);
    }
}

__global__ __launch_bounds__(256)
void bn_apply_out_k(const float* __restrict__ y, const float* __restrict__ mean,
                    const float* __restrict__ rstd, const float* __restrict__ g,
                    const float* __restrict__ be, float* __restrict__ out,
                    int cbase) {
    const int c  = blockIdx.y;
    const int p  = (blockIdx.x * 256 + threadIdx.x) * 4;
    const float sc = rstd[c] * g[c];
    const float sh = be[c] - mean[c] * sc;
    float4 v = *(const float4*)(y + (size_t)c * P_TOT + p);
    v.x = fmaxf(v.x * sc + sh, 0.f);
    v.y = fmaxf(v.y * sc + sh, 0.f);
    v.z = fmaxf(v.z * sc + sh, 0.f);
    v.w = fmaxf(v.w * sc + sh, 0.f);
    *(float4*)(out + (((size_t)(p >> 10)) * 256 + cbase + c) * 1024 + (p & 1023)) = v;
}

__global__ __launch_bounds__(256)
void bn_apply_act_k(const float* __restrict__ y, const float* __restrict__ mean,
                    const float* __restrict__ rstd, const float* __restrict__ g,
                    const float* __restrict__ be, s8* __restrict__ aout,
                    int Ci, int Cpad) {
    __shared__ __align__(16) s8 sm[32 * 128];
    const int p0 = blockIdx.x * 32;
    const int E  = 32 * Cpad;
    for (int e = threadIdx.x; e < E; e += 256) {
        int c = e >> 5, i = e & 31;
        s8 q = 0;
        if (c < Ci) {
            float v = (y[(size_t)c * P_TOT + p0 + i] - mean[c]) * rstd[c] * g[c] + be[c];
            q = q8(fmaxf(v, 0.f));
        }
        sm[i * Cpad + c] = q;
    }
    __syncthreads();
    int* dst = (int*)(aout + (size_t)p0 * Cpad);
    for (int e = threadIdx.x; e < (E >> 2); e += 256) dst[e] = ((const int*)sm)[e];
}

extern "C" void kernel_launch(void* const* d_in, const int* in_sizes, int n_in,
                              void* d_out, int out_size, void* d_ws, size_t ws_size,
                              hipStream_t stream) {
    (void)in_sizes; (void)n_in; (void)out_size; (void)ws_size;
    const float* x = (const float*)d_in[0];
    const float *b1_w = (const float*)d_in[1],  *b1_g = (const float*)d_in[3],  *b1_be = (const float*)d_in[4];
    const float *b2a_w= (const float*)d_in[5],  *b2a_g= (const float*)d_in[7],  *b2a_be= (const float*)d_in[8];
    const float *b2b_w= (const float*)d_in[9],  *b2b_g= (const float*)d_in[11], *b2b_be= (const float*)d_in[12];
    const float *b3a_w= (const float*)d_in[13], *b3a_g= (const float*)d_in[15], *b3a_be= (const float*)d_in[16];
    const float *b3b_w= (const float*)d_in[17], *b3b_g= (const float*)d_in[19], *b3b_be= (const float*)d_in[20];
    const float *b3c_w= (const float*)d_in[21], *b3c_g= (const float*)d_in[23], *b3c_be= (const float*)d_in[24];
    const float *b4_w = (const float*)d_in[25], *b4_g = (const float*)d_in[27], *b4_be = (const float*)d_in[28];
    float* out = (float*)d_out;

    char* ws = (char*)d_ws;
    size_t off = 0;
    auto carve = [&](size_t b) { size_t o = off; off = (off + b + 255) & ~(size_t)255; return ws + o; };
    s8*    xq    = (s8*)carve((size_t)P_TOT * 192);
    s8*    xpq   = (s8*)carve((size_t)P_TOT * 192);
    float* ybuf  = (float*)carve((size_t)128 * P_TOT * 4);
    s8*    act2a = (s8*)carve((size_t)P_TOT * 128);
    s8*    act3  = (s8*)carve((size_t)P_TOT * 64);
    s8*    wq1   = (s8*)carve(64 * 192);
    s8*    wq2a  = (s8*)carve(96 * 192);
    s8*    wq2b  = (s8*)carve(9 * 128 * 128);
    s8*    wq3a  = (s8*)carve(16 * 192);
    s8*    wq3b  = (s8*)carve(9 * 32 * 64);
    s8*    wq3c  = (s8*)carve(9 * 32 * 64);
    s8*    wq4   = (s8*)carve(32 * 192);
    float* meanb = (float*)carve(128 * 4);
    float* rstdb = (float*)carve(128 * 4);

    auto qw = [&](const float* w, s8* dst, int Co, int Ci, int KH, int Cpad) {
        int total = KH * KH * Co * Cpad;
        quant_w_k<<<(total + 255) / 256, 256, 0, stream>>>(w, dst, Co, Ci, KH, Cpad, total);
    };
    qw(b1_w,  wq1,  64, 192, 1, 192);
    qw(b2a_w, wq2a, 96, 192, 1, 192);
    qw(b2b_w, wq2b, 128, 96, 3, 128);
    qw(b3a_w, wq3a, 16, 192, 1, 192);
    qw(b3b_w, wq3b, 32, 16, 3, 64);
    qw(b3c_w, wq3c, 32, 32, 3, 64);
    qw(b4_w,  wq4,  32, 192, 1, 192);

    quant_x_k<<<P_TOT / 32, 256, 0, stream>>>(x, xq);
    pool_q_k <<<P_TOT / 32, 256, 0, stream>>>(x, xpq);

    auto stats = [&](int Co) { bn_stats_k<<<Co, 256, 0, stream>>>(ybuf, meanb, rstdb); };
    auto to_out = [&](int Co, const float* g, const float* be, int cbase) {
        bn_apply_out_k<<<dim3(P_TOT / 1024, Co), 256, 0, stream>>>(ybuf, meanb, rstdb, g, be, out, cbase);
    };
    auto to_act = [&](const float* g, const float* be, s8* a, int Ci, int Cpad) {
        bn_apply_act_k<<<P_TOT / 32, 256, 0, stream>>>(ybuf, meanb, rstdb, g, be, a, Ci, Cpad);
    };

    const dim3 blk(256);
    conv_wmma_iu8<1, 192><<<dim3(P_TOT / 256, 4), blk, 0, stream>>>(xq, wq1, ybuf, 64);
    stats(64);  to_out(64, b1_g, b1_be, 0);
    conv_wmma_iu8<1, 192><<<dim3(P_TOT / 256, 6), blk, 0, stream>>>(xq, wq2a, ybuf, 96);
    stats(96);  to_act(b2a_g, b2a_be, act2a, 96, 128);
    conv_wmma_iu8<9, 128><<<dim3(P_TOT / 256, 8), blk, 0, stream>>>(act2a, wq2b, ybuf, 128);
    stats(128); to_out(128, b2b_g, b2b_be, 64);
    conv_wmma_iu8<1, 192><<<dim3(P_TOT / 256, 1), blk, 0, stream>>>(xq, wq3a, ybuf, 16);
    stats(16);  to_act(b3a_g, b3a_be, act3, 16, 64);
    conv_wmma_iu8<9, 64><<<dim3(P_TOT / 256, 2), blk, 0, stream>>>(act3, wq3b, ybuf, 32);
    stats(32);  to_act(b3b_g, b3b_be, act3, 32, 64);
    conv_wmma_iu8<9, 64><<<dim3(P_TOT / 256, 2), blk, 0, stream>>>(act3, wq3c, ybuf, 32);
    stats(32);  to_out(32, b3c_g, b3c_be, 192);
    conv_wmma_iu8<1, 192><<<dim3(P_TOT / 256, 2), blk, 0, stream>>>(xpq, wq4, ybuf, 32);
    stats(32);  to_out(32, b4_g, b4_be, 224);
}
